// GAT_RCA_50070728737103
// MI455X (gfx1250) — matrix operand order repaired and checked by reading, not compiled, not run on hardware
//
#include <hip/hip_runtime.h>
#include <hip/hip_bf16.h>
#include <math.h>

// ---------------- problem constants ----------------
#define NNODES 50000
#define NPAD   50048               // NNODES rounded up to multiple of 128 (GEMM tile)
#define NEDGES 800000
#define ETOT   (NEDGES + NNODES)   // edges + self loops
#define NHEADS 4
#define OUTCH  10
#define NEG_SLOPE 0.2f

typedef _Float16 v8h  __attribute__((ext_vector_type(8)));
typedef _Float16 v16h __attribute__((ext_vector_type(16)));
typedef float    v8f  __attribute__((ext_vector_type(8)));

// ---------------- helpers ----------------
__device__ __forceinline__ unsigned enc_f32(float f) {
  int i = __float_as_int(f);
  return (i >= 0) ? ((unsigned)i | 0x80000000u) : ~(unsigned)i;
}
__device__ __forceinline__ float dec_f32(unsigned u) {
  int i = (u & 0x80000000u) ? (int)(u ^ 0x80000000u) : ~(int)u;
  return __int_as_float(i);
}
__device__ __forceinline__ void edge_sd(const long long* __restrict__ ei, int e,
                                        int& s, int& d) {
  if (e < NEDGES) { s = (int)ei[e]; d = (int)ei[NEDGES + e]; }
  else            { s = e - NEDGES; d = e - NEDGES; }          // self loop
}

// ---------------- generic elementwise kernels ----------------
// cast f32 -> f16, zero-filling the padded tail rows
__global__ void cast_f32_to_f16_pad(const float* __restrict__ in,
                                    _Float16* __restrict__ out, int n, int npad) {
  int t = blockIdx.x * blockDim.x + threadIdx.x;
  for (; t < npad; t += gridDim.x * blockDim.x)
    out[t] = (t < n) ? (_Float16)in[t] : (_Float16)0.0f;
}

__global__ void zero_f32(float* __restrict__ p, int n) {
  int t = blockIdx.x * blockDim.x + threadIdx.x;
  for (; t < n; t += gridDim.x * blockDim.x) p[t] = 0.0f;
}

__global__ void seg_init(unsigned* __restrict__ smax, float* __restrict__ ssum, int n) {
  int t = blockIdx.x * blockDim.x + threadIdx.x;
  if (t < n) { smax[t] = enc_f32(-3.402823466e38f); ssum[t] = 0.0f; }
}

// W[256][256] row-major f32  ->  Wt[col][k] f16 (column-major of W)
__global__ void wcast_transpose(const float* __restrict__ W, _Float16* __restrict__ Wt) {
  int t = blockIdx.x * blockDim.x + threadIdx.x;   // 65536 threads
  int k = t >> 8, n = t & 255;
  Wt[(size_t)n * 256 + k] = (_Float16)W[t];
}

// ---------------- WMMA GEMM:  C[NPAD x 256] = A[NPAD x 256] * B[256 x 256] --------
// A: f16 row-major (rows padded/zero-filled to NPAD); Bt: f16 column-major with two
// extra 16x256 tiles of slack for the dangling prefetches; C: f32 row-major.
// Block = 256 threads (8 waves). Wave w owns rows blockIdx.x*128 + 16*w .. +15.
// Branch-free; EXEC is all-ones at every WMMA. Each wave works on TWO column tiles
// with interleaved wmma/load so both B buffers stay live -> true load/compute overlap.
__global__ void gemm_wmma_f16_256(const _Float16* __restrict__ A,
                                  const _Float16* __restrict__ Bt,
                                  float* __restrict__ C) {
  const int wave = threadIdx.x >> 5;
  const int lane = threadIdx.x & 31;
  const int hsel = lane >> 4;          // which K-half this lane carries
  const int ln16 = lane & 15;
  const size_t rowA = (size_t)blockIdx.x * 128 + wave * 16 + ln16;

  // Preload all 8 A fragments (K = 256, 32 per WMMA).  16-bit A 16x32 (matches B):
  // lane(ln16,hsel): K = kk*32 + hsel*16 + j
  v16h a[8];
#pragma unroll
  for (int kk = 0; kk < 8; ++kk) {
    const _Float16* ap = A + rowA * 256 + kk * 32 + hsel * 16;
    v8h lo = *(const v8h*)(ap);
    v8h hi = *(const v8h*)(ap + 8);
    a[kk] = __builtin_shufflevector(lo, hi, 0,1,2,3,4,5,6,7,8,9,10,11,12,13,14,15);
  }

  const size_t rbase = (size_t)blockIdx.x * 128 + wave * 16 + hsel * 8;

  // B fragment base: 16-bit B 32x16 layout: lane(ln16,hsel) holds column ln16 of the
  // tile, K = kk*32 + hsel*16 + j  (16 contiguous halves in column-major Bt).
  const _Float16* bbase = Bt + (size_t)ln16 * 256 + hsel * 16;

  // two live B buffers (tile0 / tile1 of the current column-tile pair)
  v8h b0lo = *(const v8h*)(bbase);
  v8h b0hi = *(const v8h*)(bbase + 8);
  v8h b1lo = *(const v8h*)(bbase + 16 * 256);
  v8h b1hi = *(const v8h*)(bbase + 16 * 256 + 8);

#pragma unroll 1
  for (int ctp = 0; ctp < 8; ++ctp) {
    const _Float16* bp0 = bbase + (size_t)ctp * (32 * 256);
    const _Float16* bp1 = bp0 + 16 * 256;
    v8f acc0 = {}, acc1 = {};
#pragma unroll
    for (int kk = 0; kk < 8; ++kk) {
      // ---- tile 0: consume current fragment, immediately refill for kk+1 ----
      v16h b0 = __builtin_shufflevector(b0lo, b0hi, 0,1,2,3,4,5,6,7,8,9,10,11,12,13,14,15);
      const _Float16* n0 = (kk < 7) ? (bp0 + (kk + 1) * 32) : (bp0 + 32 * 256);
      b0lo = *(const v8h*)(n0);                 // (ctp==7,kk==7 reads Wt slack tile)
      b0hi = *(const v8h*)(n0 + 8);
      acc0 = __builtin_amdgcn_wmma_f32_16x16x32_f16(false, a[kk], false, b0,
                                                    (short)0, acc0, false, false);
      // ---- tile 1: consume current fragment, immediately refill for kk+1 ----
      v16h b1 = __builtin_shufflevector(b1lo, b1hi, 0,1,2,3,4,5,6,7,8,9,10,11,12,13,14,15);
      const _Float16* n1 = (kk < 7) ? (bp1 + (kk + 1) * 32) : (bp1 + 32 * 256);
      b1lo = *(const v8h*)(n1);                 // (ctp==7,kk==7 reads Wt slack tile)
      b1hi = *(const v8h*)(n1 + 8);
      acc1 = __builtin_amdgcn_wmma_f32_16x16x32_f16(false, a[kk], false, b1,
                                                    (short)0, acc1, false, false);
    }
    // f32 C/D layout: lane col = ln16, VGPR r -> row rbase + r  (rows padded: no guard)
    float* cp0 = C + rbase * 256 + ctp * 32 + ln16;
#pragma unroll
    for (int r = 0; r < 8; ++r) cp0[(size_t)r * 256] = acc0[r];
#pragma unroll
    for (int r = 0; r < 8; ++r) cp0[(size_t)r * 256 + 16] = acc1[r];
  }
}

// ---------------- attention ----------------
// per-node, per-head dots with a_src / a_dst
__global__ void alphas_kernel(const float* __restrict__ Hf,
                              const float* __restrict__ a_src,
                              const float* __restrict__ a_dst,
                              float* __restrict__ as, float* __restrict__ ad) {
  int t = blockIdx.x * blockDim.x + threadIdx.x;
  if (t >= NNODES * NHEADS) return;
  int node = t >> 2, h = t & 3;
  const float* hp = Hf + (size_t)node * 256 + h * 64;
  const float* sp = a_src + h * 64;
  const float* dp = a_dst + h * 64;
  float s = 0.f, d = 0.f;
#pragma unroll 4
  for (int c = 0; c < 64; ++c) { float v = hp[c]; s += v * sp[c]; d += v * dp[c]; }
  as[t] = s; ad[t] = d;
}

// leaky-relu logit + segment max (encoded u32 atomicMax)
__global__ void edge_logit_max(const long long* __restrict__ ei,
                               const float* __restrict__ as,
                               const float* __restrict__ ad,
                               float* __restrict__ ebuf,
                               unsigned* __restrict__ smax) {
  int t = blockIdx.x * blockDim.x + threadIdx.x;
  if (t >= ETOT * NHEADS) return;
  int e = t >> 2, h = t & 3;
  int s, d; edge_sd(ei, e, s, d);
  float v = as[s * 4 + h] + ad[d * 4 + h];
  v = (v > 0.f) ? v : NEG_SLOPE * v;
  ebuf[t] = v;
  atomicMax(&smax[d * 4 + h], enc_f32(v));
}

// exp(e - max) + segment sum
__global__ void edge_exp_sum(const long long* __restrict__ ei,
                             float* __restrict__ ebuf,
                             const unsigned* __restrict__ smax,
                             float* __restrict__ ssum) {
  int t = blockIdx.x * blockDim.x + threadIdx.x;
  if (t >= ETOT * NHEADS) return;
  int e = t >> 2, h = t & 3;
  int s, d; edge_sd(ei, e, s, d);
  float m = dec_f32(smax[d * 4 + h]);
  float ex = expf(ebuf[t] - m);
  ebuf[t] = ex;
  atomicAdd(&ssum[d * 4 + h], ex);
}

// one block (256 thr) per edge: out[dst] += att * h[src]  (4 heads x 64 ch)
__global__ void aggregate_kernel(const long long* __restrict__ ei,
                                 const float* __restrict__ Hf,
                                 const float* __restrict__ ebuf,
                                 const float* __restrict__ ssum,
                                 float* __restrict__ out) {
  __shared__ int sh[2];
  int e = blockIdx.x;
  if (threadIdx.x == 0) { int s, d; edge_sd(ei, e, s, d); sh[0] = s; sh[1] = d; }
  __syncthreads();
  int s = sh[0], d = sh[1];
  int h = threadIdx.x >> 6, c = threadIdx.x & 63;
  float att = ebuf[(size_t)e * 4 + h] / (ssum[d * 4 + h] + 1e-16f);
  atomicAdd(out + (size_t)d * 256 + h * 64 + c,
            Hf[(size_t)s * 256 + h * 64 + c] * att);
}

// ---------------- epilogues ----------------
// layer1: +bias, ELU, write f16 (input of layer-2 GEMM); zero pad rows
__global__ void finalize_concat(const float* __restrict__ acc,
                                const float* __restrict__ bias,
                                _Float16* __restrict__ h16) {
  int t = blockIdx.x * blockDim.x + threadIdx.x;
  if (t >= NPAD * 256) return;
  if (t < NNODES * 256) {
    float v = acc[t] + bias[t & 255];
    v = (v > 0.f) ? v : expm1f(v);
    h16[t] = (_Float16)v;
  } else {
    h16[t] = (_Float16)0.0f;
  }
}

// layer2: mean over 4 heads, +bias, ELU -> h2[N x 64] f32
__global__ void finalize_mean(const float* __restrict__ acc,
                              const float* __restrict__ bias,
                              float* __restrict__ h2) {
  int t = blockIdx.x * blockDim.x + threadIdx.x;
  if (t >= NNODES * 64) return;
  int n = t >> 6, c = t & 63;
  const float* p = acc + (size_t)n * 256 + c;
  float v = 0.25f * (p[0] + p[64] + p[128] + p[192]) + bias[c];
  v = (v > 0.f) ? v : expm1f(v);
  h2[t] = v;
}

// out[N x 10] = h2[N x 64] @ Wl[64 x 10] + bl
__global__ void head_linear(const float* __restrict__ h2,
                            const float* __restrict__ Wl,
                            const float* __restrict__ bl,
                            float* __restrict__ out) {
  int n = blockIdx.x * blockDim.x + threadIdx.x;
  if (n >= NNODES) return;
  float acc[OUTCH];
#pragma unroll
  for (int o = 0; o < OUTCH; ++o) acc[o] = bl[o];
  const float* hp = h2 + (size_t)n * 64;
#pragma unroll 4
  for (int c = 0; c < 64; ++c) {
    float hv = hp[c];
#pragma unroll
    for (int o = 0; o < OUTCH; ++o) acc[o] += hv * Wl[c * OUTCH + o];
  }
#pragma unroll
  for (int o = 0; o < OUTCH; ++o) out[(size_t)n * OUTCH + o] = acc[o];
}

// ---------------- launch ----------------
extern "C" void kernel_launch(void* const* d_in, const int* in_sizes, int n_in,
                              void* d_out, int out_size, void* d_ws, size_t ws_size,
                              hipStream_t stream) {
  const float*     x   = (const float*)d_in[0];
  const long long* ei  = (const long long*)d_in[1];   // int64 [2][E]
  const float*     W1  = (const float*)d_in[2];
  const float*     as1 = (const float*)d_in[3];
  const float*     ad1 = (const float*)d_in[4];
  const float*     b1  = (const float*)d_in[5];
  const float*     W2  = (const float*)d_in[6];
  const float*     as2 = (const float*)d_in[7];
  const float*     ad2 = (const float*)d_in[8];
  const float*     b2  = (const float*)d_in[9];
  const float*     Wl  = (const float*)d_in[10];
  const float*     bl  = (const float*)d_in[11];
  float* out = (float*)d_out;

  // ---- workspace carve-up (all offsets 256B aligned) ----
  char*  base = (char*)d_ws;
  size_t off  = 0;
  auto carve = [&](size_t bytes) { void* p = base + off;
                                   off = (off + bytes + 255) & ~(size_t)255; return p; };
  _Float16* h16   = (_Float16*)carve((size_t)NPAD * 256 * 2);   // x (f16) / layer1 act
  _Float16* W1t   = (_Float16*)carve((256 + 32) * 256 * 2);     // + 2 tiles prefetch slack
  _Float16* W2t   = (_Float16*)carve((256 + 32) * 256 * 2);     // + 2 tiles prefetch slack
  float*    Hf    = (float*)   carve((size_t)NPAD * 256 * 4);   // x@W (per layer)
  float*    acc   = (float*)   carve((size_t)NNODES * 256 * 4); // aggregation target
  float*    as    = (float*)   carve((size_t)NNODES * 4 * 4);
  float*    ad    = (float*)   carve((size_t)NNODES * 4 * 4);
  unsigned* smax  = (unsigned*)carve((size_t)NNODES * 4 * 4);
  float*    ssum  = (float*)   carve((size_t)NNODES * 4 * 4);
  float*    ebuf  = (float*)   carve((size_t)ETOT   * 4 * 4);
  float*    h2    = (float*)   carve((size_t)NNODES * 64 * 4);
  (void)ws_size; (void)in_sizes; (void)n_in; (void)out_size;

  const int TB = 256;
  const int gElem    = (NNODES * 256 + TB - 1) / TB;
  const int gElemPad = (NPAD   * 256 + TB - 1) / TB;
  const int gNH      = (NNODES * NHEADS + TB - 1) / TB;
  const int gEdgeH   = (ETOT * NHEADS + TB - 1) / TB;
  const int gGemm    = NPAD / 128;

  // prep: casts + weight transposes
  cast_f32_to_f16_pad<<<gElemPad, TB, 0, stream>>>(x, h16, NNODES * 256, NPAD * 256);
  wcast_transpose<<<256, 256, 0, stream>>>(W1, W1t);
  wcast_transpose<<<256, 256, 0, stream>>>(W2, W2t);

  // ---------------- GAT layer 1 (concat heads) ----------------
  gemm_wmma_f16_256<<<gGemm, 256, 0, stream>>>(h16, W1t, Hf);
  alphas_kernel<<<gNH, TB, 0, stream>>>(Hf, as1, ad1, as, ad);
  seg_init<<<gNH, TB, 0, stream>>>(smax, ssum, NNODES * NHEADS);
  zero_f32<<<gElem, TB, 0, stream>>>(acc, NNODES * 256);
  edge_logit_max<<<gEdgeH, TB, 0, stream>>>(ei, as, ad, ebuf, smax);
  edge_exp_sum<<<gEdgeH, TB, 0, stream>>>(ei, ebuf, smax, ssum);
  aggregate_kernel<<<ETOT, 256, 0, stream>>>(ei, Hf, ebuf, ssum, acc);
  finalize_concat<<<gElemPad, TB, 0, stream>>>(acc, b1, h16);  // h16 = elu layer1 out

  // ---------------- GAT layer 2 (mean heads) ----------------
  gemm_wmma_f16_256<<<gGemm, 256, 0, stream>>>(h16, W2t, Hf);
  alphas_kernel<<<gNH, TB, 0, stream>>>(Hf, as2, ad2, as, ad);
  seg_init<<<gNH, TB, 0, stream>>>(smax, ssum, NNODES * NHEADS);
  zero_f32<<<gElem, TB, 0, stream>>>(acc, NNODES * 256);
  edge_logit_max<<<gEdgeH, TB, 0, stream>>>(ei, as, ad, ebuf, smax);
  edge_exp_sum<<<gEdgeH, TB, 0, stream>>>(ei, ebuf, smax, ssum);
  aggregate_kernel<<<ETOT, 256, 0, stream>>>(ei, Hf, ebuf, ssum, acc);
  finalize_mean<<<(NNODES * 64 + TB - 1) / TB, TB, 0, stream>>>(acc, b2, h2);

  // ---------------- output head ----------------
  head_linear<<<(NNODES + 127) / 128, 128, 0, stream>>>(h2, Wl, bl, out);
}
